// TCL_Pool_19971597926997
// MI455X (gfx1250) — compile-verified, matrix operand order checked by reading, not run on hardware
//
#include <hip/hip_runtime.h>
#include <math.h>

// CDNA5 (gfx1250) fused TCL/PoolFormer temporal-graph kernel, round 4.
// Same algorithm as round 3 (all GEMMs via v_wmma_f32_16x16x32_bf16 on static
// LDS panels), but all helpers are forceinline free functions with explicit
// args — no [&] lambda captures, so no scratch-homed pointers / flat chases.

typedef __attribute__((ext_vector_type(16))) __bf16 v16bf;
typedef __attribute__((ext_vector_type(8)))  float  v8f;

#define NT   21
#define NTP  32
#define DM   128
#define FFD  512
#define DD   (DM*DM)
#define WKL  136
#define VLD  40
#define BSZ  4096

struct Params {
  const float* t0;
  const int*   s_nid; const int* s_eid; const float* s_nt;
  const int*   d_nid; const int* d_eid; const float* d_nt;
  const float* node_feats; const float* edge_feats;
  const float* time_w; const float* time_b; const float* depth_emb;
  const float* Wn; const float* bn; const float* We; const float* be;
  const float* Wt; const float* bt;
  const float* tf_ln1_g; const float* tf_ln1_b; const float* tf_ln2_g; const float* tf_ln2_b;
  const float* ct_ln1_g; const float* ct_ln1_b; const float* ct_ln2_g; const float* ct_ln2_b;
  const float* tf_w1; const float* tf_b1; const float* tf_w2; const float* tf_b2;
  const float* ct_wq; const float* ct_bq; const float* ct_wk; const float* ct_bk;
  const float* ct_wv; const float* ct_bv; const float* ct_wo; const float* ct_bo;
  const float* ct_w1; const float* ct_b1; const float* ct_w2; const float* ct_b2;
  const float* Wout; const float* bout;
  float* out;
};

// per-thread WMMA tiling indices (all SSA values, passed by value)
struct Ix { int tid, wid, lane, kh, row0, nb, mrow, ncol, hi8, c0l; };

static __device__ __forceinline__ v8f wmma_bf16(v16bf a, v16bf b, v8f c) {
  return __builtin_amdgcn_wmma_f32_16x16x32_bf16(false, a, false, b, (short)0, c, false, false);
}

static __device__ __forceinline__ v16bf load_frag(const __bf16* q) {
  v16bf f;
#pragma unroll
  for (int j = 0; j < 8; ++j) { f[j] = q[j]; f[8+j] = q[16+j]; }
  return f;
}

// acc[0..1] += A[32xK] @ B[Kx32 slab]; A row-major, B transposed [n][k]
static __device__ __forceinline__ void run_mm(Ix ix, const __bf16* A, int lda,
                                              const __bf16* Bt, int ldb, int K, v8f* acc) {
  const __bf16* abase  = A  + ix.mrow*lda + ix.kh;
  const __bf16* b0base = Bt + ix.ncol*ldb + ix.kh;
  const __bf16* b1base = b0base + 16*ldb;
#pragma unroll 4
  for (int k0 = 0; k0 < K; k0 += 32) {
    v16bf a  = load_frag(abase  + k0);
    v16bf b0 = load_frag(b0base + k0);
    v16bf b1 = load_frag(b1base + k0);
    acc[0] = wmma_bf16(a, b0, acc[0]);
    acc[1] = wmma_bf16(a, b1, acc[1]);
  }
}

static __device__ __forceinline__ void stage_af(int tid, __bf16* Abf, const float* X) {
  for (int i = tid; i < NTP*DM; i += 256) {
    int r = i >> 7;
    Abf[i] = (__bf16)((r < NT) ? X[i] : 0.f);
  }
}
// coalesced global reads (n fast), transposed LDS panel [n][k]
static __device__ __forceinline__ void stage_wT(int tid, __bf16* Wbf,
                                                const float* Wg, int rows) {
  for (int i = tid; i < DD; i += 256) {
    int k = i >> 7, n = i & 127;
    Wbf[n*WKL + k] = (__bf16)((k < rows) ? Wg[k*DM + n] : 0.f);
  }
}
static __device__ __forceinline__ void stage_w1T(int tid, __bf16* Wbf,
                                                 const float* W1, int c0w) {
  for (int i = tid; i < DD; i += 256) {
    int k = i >> 7, n = i & 127;
    Wbf[n*WKL + k] = (__bf16)W1[k*FFD + c0w + n];
  }
}
static __device__ __forceinline__ void stage_w2T(int tid, __bf16* Wbf, const float* W2c) {
  for (int i = tid; i < DD; i += 256) {
    int k = i >> 7, n = i & 127;
    Wbf[n*WKL + k] = (__bf16)W2c[k*DM + n];
  }
}

static __device__ __forceinline__ void ln_rows(int wid, int lane, const float* X, float* Z,
                                               const float* g, const float* bb) {
  for (int r = wid; r < NT; r += 8) {
    float v0[4]; float s = 0.f;
#pragma unroll
    for (int j = 0; j < 4; ++j) { v0[j] = X[r*DM + lane + 32*j]; s += v0[j]; }
#pragma unroll
    for (int m = 16; m >= 1; m >>= 1) s += __shfl_xor(s, m, 32);
    float mu = s * (1.f/128.f);
    float vs = 0.f;
#pragma unroll
    for (int j = 0; j < 4; ++j) { float d = v0[j]-mu; vs += d*d; }
#pragma unroll
    for (int m = 16; m >= 1; m >>= 1) vs += __shfl_xor(vs, m, 32);
    float inv = rsqrtf(vs*(1.f/128.f) + 1e-5f);
#pragma unroll
    for (int j = 0; j < 4; ++j) {
      int c = lane + 32*j;
      Z[r*DM + c] = (v0[j]-mu)*inv*g[c] + bb[c];
    }
  }
}

// out = resid + act(Xin@W1 + B1)@W2 + B2, FF chunked by 128 cols
static __device__ __forceinline__ void ffn(Ix ix, __bf16* Abf, __bf16* Hbf, __bf16* Wbf,
                                           const float* Xin,
                                           const float* W1, const float* B1,
                                           const float* W2, const float* B2,
                                           float* outp, const float* resid, int relu) {
  const v8f vz = {0.f,0.f,0.f,0.f,0.f,0.f,0.f,0.f};
  __syncthreads();
  stage_af(ix.tid, Abf, Xin);
  v8f yacc[2] = {vz, vz};
  for (int ch = 0; ch < 4; ++ch) {
    __syncthreads();
    stage_w1T(ix.tid, Wbf, W1, ch*128);
    __syncthreads();
    v8f hacc[2] = {vz, vz};
    run_mm(ix, Abf, DM, Wbf, WKL, DM, hacc);
#pragma unroll
    for (int t = 0; t < 2; ++t) {
      int c = ix.nb + t*16 + ix.c0l;
#pragma unroll
      for (int r = 0; r < 8; ++r) {
        int m = ix.row0 + r + ix.hi8;
        float v = hacc[t][r] + B1[ch*128 + c];
        v = relu ? fmaxf(v, 0.f)
                 : 0.5f*v*(1.f + tanhf(0.7978845608f*(v + 0.044715f*v*v*v)));
        Hbf[m*DM + c] = (__bf16)((m < NT) ? v : 0.f);
      }
    }
    __syncthreads();
    stage_w2T(ix.tid, Wbf, W2 + ch*128*DM);
    __syncthreads();
    run_mm(ix, Hbf, DM, Wbf, WKL, DM, yacc);
  }
  __syncthreads();
#pragma unroll
  for (int t = 0; t < 2; ++t) {
    int c = ix.nb + t*16 + ix.c0l;
#pragma unroll
    for (int r = 0; r < 8; ++r) {
      int m = ix.row0 + r + ix.hi8;
      if (m < NT) outp[m*DM + c] = resid[m*DM + c] + yacc[t][r] + B2[c];
    }
  }
  __syncthreads();
}

// fuse: gathers + 3 WMMA projections + depth emb
static __device__ __forceinline__ void fuse(Ix ix, int b, float t0v,
    __bf16* Abf, __bf16* Wbf, int* gidN, int* gidE, float* sdt,
    const int* nid, const int* eid, const float* nt_,
    const float* node_feats, const float* edge_feats,
    const float* time_w, const float* time_b, const float* depth_emb,
    const float* Wn, const float* bn, const float* We, const float* be,
    const float* Wt, const float* bt, float* X) {
  const v8f vz = {0.f,0.f,0.f,0.f,0.f,0.f,0.f,0.f};
  v8f acc[2] = {vz, vz};
  __syncthreads();
  if (ix.tid < NT) {
    gidN[ix.tid] = nid[b*NT + ix.tid];
    gidE[ix.tid] = eid[b*NT + ix.tid];
    sdt[ix.tid]  = t0v - nt_[b*NT + ix.tid];
  }
  __syncthreads();
  for (int i = ix.tid; i < NTP*DM; i += 256) {
    int r = i >> 7, c = i & 127;
    Abf[i] = (__bf16)((r < NT) ? node_feats[(long)gidN[r]*DM + c] : 0.f);
  }
  stage_wT(ix.tid, Wbf, Wn, 128);
  __syncthreads();
  run_mm(ix, Abf, DM, Wbf, WKL, DM, acc);
  __syncthreads();
  for (int i = ix.tid; i < NTP*DM; i += 256) {
    int r = i >> 7, c = i & 127;
    Abf[i] = (__bf16)((r < NT) ? edge_feats[(long)gidE[r]*DM + c] : 0.f);
  }
  stage_wT(ix.tid, Wbf, We, 128);
  __syncthreads();
  run_mm(ix, Abf, DM, Wbf, WKL, DM, acc);
  __syncthreads();
  for (int i = ix.tid; i < NTP*DM; i += 256) {
    int r = i >> 7, c = i & 127;
    float v = 0.f;
    if (r < NT && c < 100) v = __cosf(sdt[r] * time_w[c] + time_b[c]);
    Abf[i] = (__bf16)v;
  }
  stage_wT(ix.tid, Wbf, Wt, 100);
  __syncthreads();
  run_mm(ix, Abf, DM, Wbf, WKL, DM, acc);
#pragma unroll
  for (int t = 0; t < 2; ++t) {
    int c = ix.nb + t*16 + ix.c0l;
#pragma unroll
    for (int r = 0; r < 8; ++r) {
      int m = ix.row0 + r + ix.hi8;
      if (m < NT)
        X[m*DM + c] = acc[t][r] + bn[c] + be[c] + bt[c] + depth_emb[m*DM + c];
    }
  }
  __syncthreads();
}

// PoolFormer block, in place on X
static __device__ __forceinline__ void pool_block(Ix ix, float* X, float* TmA,
    __bf16* Abf, __bf16* Hbf, __bf16* Wbf,
    const float* ln1g, const float* ln1b, const float* ln2g, const float* ln2b,
    const float* W1, const float* B1, const float* W2, const float* B2) {
  ln_rows(ix.wid, ix.lane, X, TmA, ln1g, ln1b);
  __syncthreads();
  for (int i = ix.tid; i < NT*DM; i += 256) {
    int t = i >> 7, c = i & 127;
    int lo = t > 0 ? t-1 : 0, hic = t < NT-1 ? t+1 : NT-1;
    float s = 0.f;
    for (int j = lo; j <= hic; ++j) s += TmA[j*DM + c];
    X[i] += s / (float)(hic - lo + 1) - TmA[i];
  }
  __syncthreads();
  ln_rows(ix.wid, ix.lane, X, TmA, ln2g, ln2b);
  __syncthreads();
  ffn(ix, Abf, Hbf, Wbf, TmA, W1, B1, W2, B2, X, X, /*relu=*/0);
}

// post-norm cross-attention block, H=2 heads of 64
static __device__ __forceinline__ void cross(Ix ix, int l,
    const float* Xq, const float* Xkv, const int* mask, float* dest,
    float* Sb, __bf16* Abf, __bf16* Hbf, __bf16* Qbf, __bf16* Kbf,
    __bf16* Pbf, __bf16* Wbf, __bf16* Vt,
    const float* ct_wq, const float* ct_bq, const float* ct_wk, const float* ct_bk,
    const float* ct_wv, const float* ct_bv, const float* ct_wo, const float* ct_bo,
    const float* ct_ln1_g, const float* ct_ln1_b,
    const float* ct_ln2_g, const float* ct_ln2_b,
    const float* ct_w1, const float* ct_b1, const float* ct_w2, const float* ct_b2) {
  const v8f vz = {0.f,0.f,0.f,0.f,0.f,0.f,0.f,0.f};
  // Q projection -> Qbf
  __syncthreads();
  stage_af(ix.tid, Abf, Xq);
  stage_wT(ix.tid, Wbf, ct_wq + l*DD, 128);
  __syncthreads();
  { v8f acc[2] = {vz, vz};
    run_mm(ix, Abf, DM, Wbf, WKL, DM, acc);
#pragma unroll
    for (int t = 0; t < 2; ++t) { int c = ix.nb + t*16 + ix.c0l;
#pragma unroll
      for (int r = 0; r < 8; ++r) { int m = ix.row0 + r + ix.hi8;
        Qbf[m*DM + c] = (__bf16)((m < NT) ? acc[t][r] + ct_bq[l*DM + c] : 0.f); } } }
  // K projection -> Kbf
  __syncthreads();
  stage_af(ix.tid, Abf, Xkv);
  stage_wT(ix.tid, Wbf, ct_wk + l*DD, 128);
  __syncthreads();
  { v8f acc[2] = {vz, vz};
    run_mm(ix, Abf, DM, Wbf, WKL, DM, acc);
#pragma unroll
    for (int t = 0; t < 2; ++t) { int c = ix.nb + t*16 + ix.c0l;
#pragma unroll
      for (int r = 0; r < 8; ++r) { int m = ix.row0 + r + ix.hi8;
        Kbf[m*DM + c] = (__bf16)((m < NT) ? acc[t][r] + ct_bk[l*DM + c] : 0.f); } } }
  // V projection -> Vt transposed panel (Abf still holds Xkv)
  __syncthreads();
  stage_wT(ix.tid, Wbf, ct_wv + l*DD, 128);
  __syncthreads();
  { v8f acc[2] = {vz, vz};
    run_mm(ix, Abf, DM, Wbf, WKL, DM, acc);
#pragma unroll
    for (int t = 0; t < 2; ++t) { int c = ix.nb + t*16 + ix.c0l;
#pragma unroll
      for (int r = 0; r < 8; ++r) { int m = ix.row0 + r + ix.hi8;
        Vt[c*VLD + m] = (__bf16)((m < NT) ? acc[t][r] + ct_bv[l*DM + c] : 0.f); } } }
  for (int i = ix.tid; i < 2*NTP*NTP; i += 256) Pbf[i] = (__bf16)0.f;
  __syncthreads();
  // scores via WMMA: S_h = Q_h @ K_h^T (Kbf row-major IS the B^T panel)
  { int h   = ix.wid >> 2;
    int qr0 = (ix.wid & 1) << 4;
    int kc0 = ((ix.wid >> 1) & 1) << 4;
    const __bf16* qa = Qbf + (qr0 + (ix.lane & 15))*DM + h*64 + ix.kh;
    const __bf16* kb = Kbf + (kc0 + (ix.lane & 15))*DM + h*64 + ix.kh;
    v8f acc = vz;
#pragma unroll
    for (int k0 = 0; k0 < 64; k0 += 32)
      acc = wmma_bf16(load_frag(qa + k0), load_frag(kb + k0), acc);
#pragma unroll
    for (int r = 0; r < 8; ++r)
      Sb[h*1024 + (qr0 + r + ix.hi8)*32 + kc0 + ix.c0l] = acc[r];
  }
  __syncthreads();
  // masked softmax in LDS, write Pbf bf16
  if (ix.tid < 2*NT) {
    int h = ix.tid / NT, q = ix.tid - h*NT;
    float* srow = Sb + h*1024 + q*32;
    float mx = -1e30f;
    for (int k = 0; k < NT; ++k) {
      float s = mask[k] ? -1e9f : srow[k] * 0.125f;
      srow[k] = s;
      mx = fmaxf(mx, s);
    }
    float den = 0.f;
    for (int k = 0; k < NT; ++k) { float e = __expf(srow[k] - mx); srow[k] = e; den += e; }
    float inv = 1.f / den;
    for (int k = 0; k < NT; ++k) Pbf[h*NTP*NTP + q*NTP + k] = (__bf16)(srow[k]*inv);
  }
  __syncthreads();
  // O = P @ V (per-head A panel, single K=32 step) -> Abf
  { int h = (ix.wid & 3) >> 1;
    v16bf a  = load_frag(Pbf + h*NTP*NTP + ix.mrow*NTP + ix.kh);
    v16bf b0 = load_frag(Vt + ix.ncol*VLD + ix.kh);
    v16bf b1 = load_frag(Vt + (ix.ncol+16)*VLD + ix.kh);
    v8f acc[2] = {vz, vz};
    acc[0] = wmma_bf16(a, b0, acc[0]);
    acc[1] = wmma_bf16(a, b1, acc[1]);
#pragma unroll
    for (int t = 0; t < 2; ++t) { int c = ix.nb + t*16 + ix.c0l;
#pragma unroll
      for (int r = 0; r < 8; ++r) { int m = ix.row0 + r + ix.hi8;
        Abf[m*DM + c] = (__bf16)((m < NT) ? acc[t][r] : 0.f); } } }
  __syncthreads();
  // x0 = Xq + O@Wo + bo -> Sb; x1 = LN(x0) in place
  stage_wT(ix.tid, Wbf, ct_wo + l*DD, 128);
  __syncthreads();
  { v8f acc[2] = {vz, vz};
    run_mm(ix, Abf, DM, Wbf, WKL, DM, acc);
#pragma unroll
    for (int t = 0; t < 2; ++t) { int c = ix.nb + t*16 + ix.c0l;
#pragma unroll
      for (int r = 0; r < 8; ++r) { int m = ix.row0 + r + ix.hi8;
        Sb[m*DM + c] = (m < NT) ? acc[t][r] + ct_bo[l*DM + c] + Xq[m*DM + c] : 0.f; } } }
  __syncthreads();
  ln_rows(ix.wid, ix.lane, Sb, Sb, ct_ln1_g + l*DM, ct_ln1_b + l*DM);
  __syncthreads();
  ffn(ix, Abf, Hbf, Wbf, Sb, ct_w1 + l*DM*FFD, ct_b1 + l*FFD,
      ct_w2 + l*FFD*DM, ct_b2 + l*DM, Sb, Sb, /*relu=*/1);
  ln_rows(ix.wid, ix.lane, Sb, dest, ct_ln2_g + l*DM, ct_ln2_b + l*DM);
  __syncthreads();
}

static __device__ __forceinline__ void head(int tid, int b, const float* X, float* Sb,
                                            const float* Wout, const float* bout, float* o) {
  if (tid < DM) {
    float s = 0.f;
    for (int r = 0; r < NT; ++r) s += X[r*DM + tid];
    Sb[tid] = s / (float)NT;
  }
  __syncthreads();
  if (tid < DM) {
    float s = bout[tid];
    for (int k = 0; k < DM; ++k) s += Sb[k] * Wout[k*DM + tid];
    o[(long)b*DM + tid] = s;
  }
  __syncthreads();
}

__global__ __launch_bounds__(256) void tcl_fused_kernel(Params p) {
  __shared__ float  Xs [NTP*DM];
  __shared__ float  Xd [NTP*DM];
  __shared__ float  TmA[NTP*DM];
  __shared__ float  Sb [NTP*DM];
  __shared__ __bf16 Abf[NTP*DM];
  __shared__ __bf16 Hbf[NTP*DM];
  __shared__ __bf16 Qbf[NTP*DM];
  __shared__ __bf16 Kbf[NTP*DM];
  __shared__ __bf16 Pbf[2*NTP*NTP];
  __shared__ __bf16 Wbf[DM*WKL];
  __shared__ __bf16 Vt [DM*VLD];
  __shared__ int    mS[NTP], mD[NTP];
  __shared__ int    gidN[NTP], gidE[NTP];
  __shared__ float  sdt[NTP];

  const int b    = blockIdx.x;
  const int tid  = threadIdx.x;
  const int wid  = tid >> 5;
  const int lane = tid & 31;
  Ix ix;
  ix.tid  = tid;  ix.wid = wid;  ix.lane = lane;
  ix.kh   = (lane >> 4) << 3;
  ix.row0 = (wid >> 2) << 4;
  ix.nb   = (wid & 3) << 5;
  ix.mrow = ix.row0 + (lane & 15);
  ix.ncol = ix.nb + (lane & 15);
  ix.hi8  = (lane >> 4) << 3;
  ix.c0l  = lane & 15;

  if (tid < NT) {
    mS[tid] = (p.s_nid[b*NT + tid] == 0);
    mD[tid] = (p.d_nid[b*NT + tid] == 0);
  }
  const float t0v = p.t0[b];

  fuse(ix, b, t0v, Abf, Wbf, gidN, gidE, sdt,
       p.s_nid, p.s_eid, p.s_nt, p.node_feats, p.edge_feats,
       p.time_w, p.time_b, p.depth_emb,
       p.Wn, p.bn, p.We, p.be, p.Wt, p.bt, Xs);
  fuse(ix, b, t0v, Abf, Wbf, gidN, gidE, sdt,
       p.d_nid, p.d_eid, p.d_nt, p.node_feats, p.edge_feats,
       p.time_w, p.time_b, p.depth_emb,
       p.Wn, p.bn, p.We, p.be, p.Wt, p.bt, Xd);

  for (int l = 0; l < 2; ++l) {
    pool_block(ix, Xs, TmA, Abf, Hbf, Wbf,
               p.tf_ln1_g + l*DM, p.tf_ln1_b + l*DM, p.tf_ln2_g + l*DM, p.tf_ln2_b + l*DM,
               p.tf_w1 + l*DM*FFD, p.tf_b1 + l*FFD, p.tf_w2 + l*FFD*DM, p.tf_b2 + l*DM);
    pool_block(ix, Xd, TmA, Abf, Hbf, Wbf,
               p.tf_ln1_g + l*DM, p.tf_ln1_b + l*DM, p.tf_ln2_g + l*DM, p.tf_ln2_b + l*DM,
               p.tf_w1 + l*DM*FFD, p.tf_b1 + l*FFD, p.tf_w2 + l*FFD*DM, p.tf_b2 + l*DM);
    cross(ix, l, Xs, Xd, mD, TmA, Sb, Abf, Hbf, Qbf, Kbf, Pbf, Wbf, Vt,
          p.ct_wq, p.ct_bq, p.ct_wk, p.ct_bk, p.ct_wv, p.ct_bv, p.ct_wo, p.ct_bo,
          p.ct_ln1_g, p.ct_ln1_b, p.ct_ln2_g, p.ct_ln2_b,
          p.ct_w1, p.ct_b1, p.ct_w2, p.ct_b2);
    cross(ix, l, Xd, Xs, mS, Xd, Sb, Abf, Hbf, Qbf, Kbf, Pbf, Wbf, Vt,
          p.ct_wq, p.ct_bq, p.ct_wk, p.ct_bk, p.ct_wv, p.ct_bv, p.ct_wo, p.ct_bo,
          p.ct_ln1_g, p.ct_ln1_b, p.ct_ln2_g, p.ct_ln2_b,
          p.ct_w1, p.ct_b1, p.ct_w2, p.ct_b2);
    for (int i = tid; i < NT*DM; i += 256) Xs[i] = TmA[i];
    __syncthreads();
  }
  head(tid, b, Xs, Sb, p.Wout, p.bout, p.out);
  head(tid, b, Xd, Sb, p.Wout, p.bout, p.out + (long)BSZ*DM);
}

extern "C" void kernel_launch(void* const* d_in, const int* in_sizes, int n_in,
                              void* d_out, int out_size, void* d_ws, size_t ws_size,
                              hipStream_t stream) {
  (void)in_sizes; (void)n_in; (void)out_size; (void)d_ws; (void)ws_size;
  Params p;
  p.t0         = (const float*)d_in[0];
  p.s_nid      = (const int*)  d_in[1];
  p.s_eid      = (const int*)  d_in[2];
  p.s_nt       = (const float*)d_in[3];
  p.d_nid      = (const int*)  d_in[4];
  p.d_eid      = (const int*)  d_in[5];
  p.d_nt       = (const float*)d_in[6];
  p.node_feats = (const float*)d_in[7];
  p.edge_feats = (const float*)d_in[8];
  p.time_w     = (const float*)d_in[9];
  p.time_b     = (const float*)d_in[10];
  p.depth_emb  = (const float*)d_in[11];
  p.Wn = (const float*)d_in[12]; p.bn = (const float*)d_in[13];
  p.We = (const float*)d_in[14]; p.be = (const float*)d_in[15];
  p.Wt = (const float*)d_in[16]; p.bt = (const float*)d_in[17];
  p.tf_ln1_g = (const float*)d_in[18]; p.tf_ln1_b = (const float*)d_in[19];
  p.tf_ln2_g = (const float*)d_in[20]; p.tf_ln2_b = (const float*)d_in[21];
  p.ct_ln1_g = (const float*)d_in[22]; p.ct_ln1_b = (const float*)d_in[23];
  p.ct_ln2_g = (const float*)d_in[24]; p.ct_ln2_b = (const float*)d_in[25];
  p.tf_w1 = (const float*)d_in[26]; p.tf_b1 = (const float*)d_in[27];
  p.tf_w2 = (const float*)d_in[28]; p.tf_b2 = (const float*)d_in[29];
  p.ct_wq = (const float*)d_in[30]; p.ct_bq = (const float*)d_in[31];
  p.ct_wk = (const float*)d_in[32]; p.ct_bk = (const float*)d_in[33];
  p.ct_wv = (const float*)d_in[34]; p.ct_bv = (const float*)d_in[35];
  p.ct_wo = (const float*)d_in[36]; p.ct_bo = (const float*)d_in[37];
  p.ct_w1 = (const float*)d_in[38]; p.ct_b1 = (const float*)d_in[39];
  p.ct_w2 = (const float*)d_in[40]; p.ct_b2 = (const float*)d_in[41];
  p.Wout  = (const float*)d_in[42]; p.bout = (const float*)d_in[43];
  p.out   = (float*)d_out;

  tcl_fused_kernel<<<BSZ, 256, 0, stream>>>(p);
}
